// DCGRUCell_90185723281725
// MI455X (gfx1250) — matrix operand order repaired and checked by reading, not compiled, not run on hardware
//
#include <hip/hip_runtime.h>
#include <hip/hip_bf16.h>
#include <stdint.h>

// ---------------------------------------------------------------------------
// DCGRU cell forward for MI455X (gfx1250, wave32, WMMA + TDM + async-LDS).
//   B=64, N=1024, D_IN=2, U=64, K=2  ->  F=66, F*B=4224, M=5, in_sz=330
// Dominant cost: 8x bf16 GEMM [1024x1024]x[1024x4224] (f32 accum) +
// 2x projection GEMM [65536x352]x[352x{128,64}].  All GEMMs use
// v_wmma_f32_16x16x32_bf16.  A-panels staged to LDS by the Tensor Data
// Mover (tensor_load_to_lds, TENSORcnt), B-panels by
// global_load_async_to_lds_b128 (ASYNCcnt), double-buffered so DMA of
// tile t+1 overlaps WMMA on tile t.
// ---------------------------------------------------------------------------

#define B_    64
#define N_    1024
#define DIN_  2
#define U_    64
#define F_    66
#define FB_   4224          // F_*B_
#define NM_   5             // number of Chebyshev matrices
#define INSZ_ 330           // F_*NM_
#define KP_   352           // padded projection K (11*32)
#define ROWS_ 65536         // B_*N_

typedef __attribute__((ext_vector_type(16))) __bf16       bf16x16;
typedef __attribute__((ext_vector_type(8)))  float        f32x8;
typedef __attribute__((ext_vector_type(4)))  float        f32x4;
typedef __attribute__((ext_vector_type(4)))  unsigned int u32x4;
typedef __attribute__((ext_vector_type(8)))  int          i32x8;
typedef __attribute__((ext_vector_type(4)))  int          i32x4;

union Frag32 { bf16x16 v; u32x4 u[2]; };
union Pack8  { __bf16 h[8]; u32x4 u; };

__device__ __forceinline__ float sigmoidf_(float x) {
    return 1.0f / (1.0f + __expf(-x));
}

// ---------------------------------------------------------------------------
// TDM: load a [rows x 32] bf16 tile (row stride = k_stride elements) from
// global into LDS at byte offset lds_off, padding +16B after every 64B so the
// LDS row stride is 80B (= 40 halves), matching the compute-side layout.
// D# bit packing per CDNA5 ISA.  This toolchain uses
// the 6-arg builtin (g0, g1, g2, g3, g4, cpol).
// ---------------------------------------------------------------------------
__device__ __forceinline__ void tdm_load_tile(const __bf16* gaddr, unsigned lds_off,
                                              int rows, int k_stride)
{
    unsigned long long ga = (unsigned long long)(uintptr_t)gaddr;
    u32x4 g0;
    g0[0] = 1u;                                        // count=1, user descriptor
    g0[1] = lds_off;                                   // lds_addr (bytes)
    g0[2] = (unsigned)(ga & 0xffffffffu);              // global_addr[31:0]
    g0[3] = (unsigned)((ga >> 32) & 0x01ffffffu)       // global_addr[56:32]
          | (2u << 30);                                // type = 2 ("image")
    i32x8 g1;
    g1[0] = (1 << 16)                                  // data_size = 1 -> 2 bytes
          | (1 << 20)                                  // pad_enable
          | (3 << 22)                                  // pad_interval: 16 DW = 64B
          | (3 << 25);                                 // pad_amount:   4 DW = 16B
    g1[1] = (int)(32u << 16);                          // tensor_dim0[15:0] = 32
    g1[2] = (int)(((unsigned)rows & 0xffffu) << 16);   // tensor_dim1[15:0] = rows
    g1[3] = (int)(32u << 16);                          // tile_dim0 = 32
    g1[4] = (int)((unsigned)rows & 0xffffu);           // tile_dim1 = rows, tile_dim2=0
    g1[5] = k_stride;                                  // tensor_dim0_stride[31:0]
    g1[6] = 0;
    g1[7] = 0;
    i32x4 z4 = { 0, 0, 0, 0 };
    i32x8 z8 = { 0, 0, 0, 0, 0, 0, 0, 0 };
    __builtin_amdgcn_tensor_load_to_lds(g0, g1, z4, z4, z8, 0);
}

// ---------------------------------------------------------------------------
// Async per-lane DMA of a [ROWS x 32] bf16 tile into padded LDS (stride 40
// halves).  256 threads; each thread moves 16B per 64-row chunk.
// ---------------------------------------------------------------------------
template<int ROWS>
__device__ __forceinline__ void async_tile_issue(const __bf16* g, int ldg,
                                                 unsigned lds_base, int tid)
{
    const int row0 = tid >> 2;
    const int col  = (tid & 3) * 8;
#pragma unroll
    for (int r0 = 0; r0 < ROWS; r0 += 64) {
        const int row = r0 + row0;
        unsigned loff = lds_base + (unsigned)((row * 40 + col) * 2);
        const __bf16* ga = g + (size_t)row * ldg + col;
        asm volatile("global_load_async_to_lds_b128 %0, %1, off"
                     :: "v"(loff), "v"(ga) : "memory");
    }
}

// ---------------------------------------------------------------------------
// prep: fp32 supports -> bf16; weights -> transposed+padded bf16 [out, KP_]
// ---------------------------------------------------------------------------
__global__ void prep_kernel(const float* __restrict__ s0, const float* __restrict__ s1,
                            const float* __restrict__ Wo, const float* __restrict__ Wu,
                            __bf16* __restrict__ s0b, __bf16* __restrict__ s1b,
                            __bf16* __restrict__ WoT, __bf16* __restrict__ WuT)
{
    const int NT = N_ * N_;
    const int total = 2 * NT + KP_ * 128 + KP_ * 64;
    for (int i = blockIdx.x * blockDim.x + threadIdx.x; i < total;
         i += gridDim.x * blockDim.x) {
        if (i < NT) {
            s0b[i] = (__bf16)s0[i];
        } else if (i < 2 * NT) {
            s1b[i - NT] = (__bf16)s1[i - NT];
        } else if (i < 2 * NT + KP_ * 128) {
            int j = i - 2 * NT;
            int o = j / KP_, kp = j % KP_;
            WoT[o * KP_ + kp] = (kp < INSZ_) ? (__bf16)Wo[kp * 128 + o] : (__bf16)0.0f;
        } else {
            int j = i - 2 * NT - KP_ * 128;
            int o = j / KP_, kp = j % KP_;
            WuT[o * KP_ + kp] = (kp < INSZ_) ? (__bf16)Wu[kp * 64 + o] : (__bf16)0.0f;
        }
    }
}

// ---------------------------------------------------------------------------
// pack x0^T [FB_, N_]: row (f*B+b), col n = cat(inputs, hx)[b,n,f]
// ---------------------------------------------------------------------------
__global__ void pack_x0_kernel(const float* __restrict__ inputs,
                               const float* __restrict__ hx,
                               __bf16* __restrict__ X0T)
{
    int i = blockIdx.x * blockDim.x + threadIdx.x;
    if (i >= FB_ * N_) return;
    int n  = i & (N_ - 1);
    int fb = i >> 10;
    int f  = fb >> 6;       // fb / B_
    int b  = fb & 63;
    float v = (f < DIN_) ? inputs[b * (N_ * DIN_) + n * DIN_ + f]
                         : hx[b * (N_ * U_) + n * U_ + (f - DIN_)];
    X0T[i] = (__bf16)v;
}

// same, but hidden part is r*hx with r = sigmoid(y1T[0..63])
__global__ void pack_x0_gate_kernel(const float* __restrict__ inputs,
                                    const float* __restrict__ hx,
                                    const float* __restrict__ y1T,   // [128, ROWS_]
                                    __bf16* __restrict__ X0T)
{
    int i = blockIdx.x * blockDim.x + threadIdx.x;
    if (i >= FB_ * N_) return;
    int n  = i & (N_ - 1);
    int fb = i >> 10;
    int f  = fb >> 6;
    int b  = fb & 63;
    float v;
    if (f < DIN_) {
        v = inputs[b * (N_ * DIN_) + n * DIN_ + f];
    } else {
        int u   = f - DIN_;
        int row = b * N_ + n;
        float r = sigmoidf_(y1T[u * ROWS_ + row]);
        v = r * hx[b * (N_ * U_) + n * U_ + u];
    }
    X0T[i] = (__bf16)v;
}

// ---------------------------------------------------------------------------
// Unified WMMA GEMM:  C[m, j] = sum_k A[m, k] * Bm[j, k]   (A,Bm row-major in K)
//   EPI 0: outB[j*ldOut + m] = bf16(acc)                      (Y^T store)
//   EPI 1: outB[j*ldOut + m] = bf16(2*acc - X0T[j*ldOut + m]) (Chebyshev step)
//   EPI 2: outF[j*ldOut + m] = acc + bias[j]                  (projection^T)
// BM x BJ tile, BK=32, 8 waves (WAVES_M x WAVES_J), each wave MS x JS 16x16
// WMMA tiles.  Double-buffered LDS; A staged by TDM (wave 0), B by async
// per-lane loads; tile t+1 DMA overlaps WMMA on tile t.
// ---------------------------------------------------------------------------
template<int BM, int BJ, int WAVES_M, int WAVES_J, int MS, int JS, int EPI>
__global__ __launch_bounds__(256)
void wmma_gemm_nt(const __bf16* __restrict__ A, const __bf16* __restrict__ Bm,
                  int K,
                  const __bf16* __restrict__ X0T,
                  const float* __restrict__ bias,
                  __bf16* __restrict__ outB, float* __restrict__ outF,
                  int ldOut)
{
    constexpr int LDT = 40;                  // LDS row stride in halves (32 + pad)
    constexpr int ASZ = BM * LDT;            // halves per A stage
    constexpr int BSZ = BJ * LDT;            // halves per B stage
    constexpr int NB  = BJ / 64;             // async loads per thread per tile
    constexpr unsigned STAGE = (unsigned)(ASZ + BSZ) * 2u;   // stage stride bytes
    __shared__ __bf16 lds[2 * (ASZ + BSZ)];

    const int tid  = threadIdx.x;
    const int wid  = tid >> 5;
    const int lane = tid & 31;
    const int wm   = wid / WAVES_J;
    const int wj   = wid % WAVES_J;
    const int mtile = blockIdx.x * BM;
    const int jtile = blockIdx.y * BJ;
    const int half = lane >> 4;              // lane group (K split)
    const int l16  = lane & 15;
    const int kfoA = half * 8;               // A frag: K 0-7/16-23 vs 8-15/24-31

    // Runtime LDS base (kept out of any static-initialized aggregate).
    const unsigned ldsBase = (unsigned)(uintptr_t)&lds[0];

    const __bf16* Abase = A  + (size_t)mtile * K;
    const __bf16* Bbase = Bm + (size_t)jtile * K;

    f32x8 acc[MS][JS] = {};

    // prologue: stage 0 DMA
    if (wid == 0) tdm_load_tile(Abase, ldsBase, BM, K);
    async_tile_issue<BJ>(Bbase, K, ldsBase + (unsigned)ASZ * 2u, tid);

    const int T = K / 32;
    for (int t = 0; t < T; t++) {
        const int cur = t & 1;
        const unsigned nxtOff = cur ? 0u : STAGE;    // stage being filled
        __syncthreads();                     // nxt stage free (readers of t-1 done)
        if (t + 1 < T) {
            if (wid == 0)
                tdm_load_tile(Abase + (t + 1) * 32, ldsBase + nxtOff, BM, K);
            async_tile_issue<BJ>(Bbase + (t + 1) * 32, K,
                                 ldsBase + nxtOff + (unsigned)ASZ * 2u, tid);
            if (wid == 0) __builtin_amdgcn_s_wait_tensorcnt((short)1);
            asm volatile("s_wait_asynccnt %0" :: "n"(NB) : "memory");
        } else {
            if (wid == 0) __builtin_amdgcn_s_wait_tensorcnt((short)0);
            asm volatile("s_wait_asynccnt 0" ::: "memory");
        }
        __syncthreads();                     // cur stage fully in LDS

        const __bf16* As = lds + (cur ? (ASZ + BSZ) : 0);
        const __bf16* Bs = As + ASZ;
        Frag32 af[MS], bf[JS];
#pragma unroll
        for (int i = 0; i < MS; i++) {
            const __bf16* p = &As[(wm * MS * 16 + i * 16 + l16) * LDT + kfoA];
            af[i].u[0] = *(const u32x4*)p;
            af[i].u[1] = *(const u32x4*)(p + 16);
        }
#pragma unroll
        for (int j = 0; j < JS; j++) {
            // B operand: lane half selects K 0-7/16-23 vs 8-15/24-31 (matches A)
            const __bf16* p = &Bs[(wj * JS * 16 + j * 16 + l16) * LDT + half * 8];
            bf[j].u[0] = *(const u32x4*)p;
            bf[j].u[1] = *(const u32x4*)(p + 16);
        }
#pragma unroll
        for (int i = 0; i < MS; i++)
#pragma unroll
            for (int j = 0; j < JS; j++)
                acc[i][j] = __builtin_amdgcn_wmma_f32_16x16x32_bf16(
                    false, af[i].v, false, bf[j].v, (short)0, acc[i][j],
                    false, false);
    }

    // Epilogue.  D layout: VGPR v -> M = v + 8*half, N(col j) = l16.
#pragma unroll
    for (int i = 0; i < MS; i++) {
#pragma unroll
        for (int j = 0; j < JS; j++) {
            const int jg = jtile + wj * JS * 16 + j * 16 + l16;
            const int mg = mtile + wm * MS * 16 + i * 16 + half * 8;
            if (EPI == 2) {
                const float bv = bias[jg];
                float* op = outF + (size_t)jg * ldOut + mg;
                f32x4 lo = { acc[i][j][0] + bv, acc[i][j][1] + bv,
                             acc[i][j][2] + bv, acc[i][j][3] + bv };
                f32x4 hi = { acc[i][j][4] + bv, acc[i][j][5] + bv,
                             acc[i][j][6] + bv, acc[i][j][7] + bv };
                *(f32x4*)op       = lo;
                *(f32x4*)(op + 4) = hi;
            } else {
                Pack8 pk;
                if (EPI == 1) {
                    Pack8 x0;
                    x0.u = *(const u32x4*)(X0T + (size_t)jg * ldOut + mg);
#pragma unroll
                    for (int v = 0; v < 8; v++)
                        pk.h[v] = (__bf16)(2.0f * acc[i][j][v] - (float)x0.h[v]);
                } else {
#pragma unroll
                    for (int v = 0; v < 8; v++)
                        pk.h[v] = (__bf16)acc[i][j][v];
                }
                *(u32x4*)(outB + (size_t)jg * ldOut + mg) = pk.u;
            }
        }
    }
}

// ---------------------------------------------------------------------------
// Gather diffusion outputs into projection A matrix:
//   Xg[(b*N+n), f*5+m] = Xm^T[(f*B+b), n];  pad cols 330..351 = 0
// ---------------------------------------------------------------------------
__global__ void gather_xg_kernel(const __bf16* __restrict__ XT,  // 5 contiguous [FB_,N_]
                                 __bf16* __restrict__ Xg)
{
    int i = blockIdx.x * blockDim.x + threadIdx.x;
    if (i >= ROWS_ * KP_) return;
    int kp  = i % KP_;
    int row = i / KP_;
    __bf16 v;
    if (kp < INSZ_) {
        int f = kp / NM_;
        int m = kp % NM_;
        int b = row >> 10;
        int n = row & (N_ - 1);
        v = XT[(size_t)m * (FB_ * N_) + (size_t)(f * B_ + b) * N_ + n];
    } else {
        v = (__bf16)0.0f;
    }
    Xg[i] = v;
}

// ---------------------------------------------------------------------------
// final: out = u*hx + (1-u)*tanh(c);  u = sigmoid(y1T[64+j]), c = y2T[j]
// ---------------------------------------------------------------------------
__global__ void final_kernel(const float* __restrict__ hx,
                             const float* __restrict__ y1T,  // [128, ROWS_]
                             const float* __restrict__ y2T,  // [ 64, ROWS_]
                             float* __restrict__ out)
{
    int i = blockIdx.x * blockDim.x + threadIdx.x;
    if (i >= B_ * N_ * U_) return;
    int j   = i & 63;
    int row = i >> 6;                 // b*N_ + n
    float u = sigmoidf_(y1T[(U_ + j) * ROWS_ + row]);
    float c = tanhf(y2T[j * ROWS_ + row]);
    out[i] = u * hx[i] + (1.0f - u) * c;
}

// ---------------------------------------------------------------------------
extern "C" void kernel_launch(void* const* d_in, const int* in_sizes, int n_in,
                              void* d_out, int out_size, void* d_ws, size_t ws_size,
                              hipStream_t stream)
{
    (void)in_sizes; (void)n_in; (void)out_size; (void)ws_size;

    const float* inputs = (const float*)d_in[0];
    const float* hx     = (const float*)d_in[1];
    const float* s0     = (const float*)d_in[2];
    const float* s1     = (const float*)d_in[3];
    const float* Wo     = (const float*)d_in[4];
    const float* bo     = (const float*)d_in[5];
    const float* Wu     = (const float*)d_in[6];
    const float* bu     = (const float*)d_in[7];
    float* out = (float*)d_out;

    char* ws = (char*)d_ws;
    size_t off = 0;
    auto alloc = [&](size_t bytes) -> void* {
        void* p = ws + off;
        off += (bytes + 255) & ~(size_t)255;
        return p;
    };

    // 5 contiguous transposed diffusion matrices: x0, S0x1, S0x2, S1x1, S1x2
    __bf16* XT0 = (__bf16*)alloc((size_t)FB_ * N_ * 2);
    __bf16* XT1 = (__bf16*)alloc((size_t)FB_ * N_ * 2);
    __bf16* XT2 = (__bf16*)alloc((size_t)FB_ * N_ * 2);
    __bf16* XT3 = (__bf16*)alloc((size_t)FB_ * N_ * 2);
    __bf16* XT4 = (__bf16*)alloc((size_t)FB_ * N_ * 2);
    __bf16* s0b = (__bf16*)alloc((size_t)N_ * N_ * 2);
    __bf16* s1b = (__bf16*)alloc((size_t)N_ * N_ * 2);
    __bf16* WoT = (__bf16*)alloc((size_t)KP_ * 128 * 2);
    __bf16* WuT = (__bf16*)alloc((size_t)KP_ * 64 * 2);
    __bf16* Xg  = (__bf16*)alloc((size_t)ROWS_ * KP_ * 2);
    float*  y1T = (float*)alloc((size_t)128 * ROWS_ * 4);
    float*  y2T = (float*)alloc((size_t)64  * ROWS_ * 4);

    const dim3 blk(256);
    const int packBlocks   = (FB_ * N_) / 256;          // 16896
    const int gatherBlocks = (ROWS_ * KP_) / 256;       // 90112
    const int finalBlocks  = (B_ * N_ * U_) / 256;      // 16384
    const dim3 gDiff(N_ / 128, FB_ / 128);              // (8, 33)
    const dim3 gP1(ROWS_ / 128, 1);                     // 128-wide proj
    const dim3 gP2(ROWS_ / 128, 1);                     // 64-wide proj

    prep_kernel<<<2048, blk, 0, stream>>>(s0, s1, Wo, Wu, s0b, s1b, WoT, WuT);

    // ---------------- gconv 1 (gates) ----------------
    pack_x0_kernel<<<packBlocks, blk, 0, stream>>>(inputs, hx, XT0);
    wmma_gemm_nt<128,128,2,4,4,2,0><<<gDiff, blk, 0, stream>>>(s0b, XT0, N_, nullptr, nullptr, XT1, nullptr, N_);
    wmma_gemm_nt<128,128,2,4,4,2,1><<<gDiff, blk, 0, stream>>>(s0b, XT1, N_, XT0,    nullptr, XT2, nullptr, N_);
    wmma_gemm_nt<128,128,2,4,4,2,0><<<gDiff, blk, 0, stream>>>(s1b, XT0, N_, nullptr, nullptr, XT3, nullptr, N_);
    wmma_gemm_nt<128,128,2,4,4,2,1><<<gDiff, blk, 0, stream>>>(s1b, XT3, N_, XT0,    nullptr, XT4, nullptr, N_);
    gather_xg_kernel<<<gatherBlocks, blk, 0, stream>>>(XT0, Xg);
    wmma_gemm_nt<128,128,2,4,4,2,2><<<gP1, blk, 0, stream>>>(Xg, WoT, KP_, nullptr, bo, nullptr, y1T, ROWS_);

    // ---------------- gconv 2 (candidate) ----------------
    pack_x0_gate_kernel<<<packBlocks, blk, 0, stream>>>(inputs, hx, y1T, XT0);
    wmma_gemm_nt<128,128,2,4,4,2,0><<<gDiff, blk, 0, stream>>>(s0b, XT0, N_, nullptr, nullptr, XT1, nullptr, N_);
    wmma_gemm_nt<128,128,2,4,4,2,1><<<gDiff, blk, 0, stream>>>(s0b, XT1, N_, XT0,    nullptr, XT2, nullptr, N_);
    wmma_gemm_nt<128,128,2,4,4,2,0><<<gDiff, blk, 0, stream>>>(s1b, XT0, N_, nullptr, nullptr, XT3, nullptr, N_);
    wmma_gemm_nt<128,128,2,4,4,2,1><<<gDiff, blk, 0, stream>>>(s1b, XT3, N_, XT0,    nullptr, XT4, nullptr, N_);
    gather_xg_kernel<<<gatherBlocks, blk, 0, stream>>>(XT0, Xg);
    wmma_gemm_nt<128,64,4,2,2,2,2><<<gP2, blk, 0, stream>>>(Xg, WuT, KP_, nullptr, bu, nullptr, y2T, ROWS_);

    // ---------------- GRU combine ----------------
    final_kernel<<<finalBlocks, blk, 0, stream>>>(hx, y1T, y2T, out);
}
